// GGNNSum_67671504716367
// MI455X (gfx1250) — matrix operand order repaired and checked by reading, not compiled, not run on hardware
//
#include <hip/hip_runtime.h>
#include <hip/hip_bf16.h>

typedef __bf16 bf16_t;
typedef __attribute__((ext_vector_type(16))) __bf16 v16bf;
typedef __attribute__((ext_vector_type(8)))  __bf16 v8bf;
typedef __attribute__((ext_vector_type(8)))  float  v8f;

#define DIM    128
#define N_ET   4
#define GDIM3  384   // 3*DIM

// ---------------------------------------------------------------- weights->bf16
__global__ void cvt_weights_kernel(const float* __restrict__ We,
                                   const float* __restrict__ Wih,
                                   const float* __restrict__ Whh,
                                   bf16_t* __restrict__ dWe,
                                   bf16_t* __restrict__ dWih,
                                   bf16_t* __restrict__ dWhh) {
    int i = blockIdx.x * blockDim.x + threadIdx.x;
    if (i < N_ET * DIM * DIM) dWe[i] = (bf16_t)We[i];
    if (i < GDIM3 * DIM) {
        dWih[i] = (bf16_t)Wih[i];
        dWhh[i] = (bf16_t)Whh[i];
    }
}

__global__ void zero_f32_kernel(float* __restrict__ p, long n) {
    long i = (long)blockIdx.x * blockDim.x + threadIdx.x;
    if (i < n) p[i] = 0.0f;
}

// Load a 16x32 bf16 A-operand: lane m=lane&15 holds row m0+m,
// K base = kk + (lane>=16 ? 16 : 0); elements 0..15 -> K+0..15.
__device__ __forceinline__ v16bf load_a_bf16(const float* __restrict__ row, int kb) {
    v16bf a;
#pragma unroll
    for (int j = 0; j < 16; ++j)
        a[j] = (bf16_t)row[kb + j];
    return a;
}

// ---------------- T[t] = h @ W[t]^T + b[t], all 4 etypes per block, bf16 output
__global__ __launch_bounds__(256)
void etype_gemm_kernel(const float* __restrict__ h, const bf16_t* __restrict__ Wt,
                       const float* __restrict__ b_et, bf16_t* __restrict__ T, int N) {
    __shared__ bf16_t st[16][DIM];              // one 16x128 bf16 tile staging area

    const int m0    = blockIdx.x * 16;
    const int wave  = threadIdx.x >> 5;
    const int lane  = threadIdx.x & 31;
    const int lmod  = lane & 15;
    const int lhalf = lane >> 4;
    const int o0    = wave * 16;                // 8 waves cover all 128 outputs

    const float* __restrict__ hrow = h + (size_t)(m0 + lmod) * DIM;

    v8f c[N_ET];
#pragma unroll
    for (int t = 0; t < N_ET; ++t) {
        float bias = b_et[t * DIM + o0 + lmod];
#pragma unroll
        for (int r = 0; r < 8; ++r) c[t][r] = bias;
    }

#pragma unroll
    for (int kk = 0; kk < DIM; kk += 32) {
        v16bf av = load_a_bf16(hrow, kk + lhalf * 16);   // A converted ONCE per block
        int kb2 = kk + lhalf * 16;
#pragma unroll
        for (int t = 0; t < N_ET; ++t) {
            // B[k][o] = W[t][o][k]; lane holds col o0+lmod
            v16bf bv = *(const v16bf*)(Wt + ((size_t)t * DIM + o0 + lmod) * DIM + kb2);
            c[t] = __builtin_amdgcn_wmma_f32_16x16x32_bf16(false, av, false, bv,
                                                           (short)0, c[t], false, false);
        }
    }

    // stage each etype tile through LDS -> coalesced 16B bf16 stores
    const int srow = threadIdx.x >> 4;          // 0..15
    const int schk = threadIdx.x & 15;          // 0..15, 8 bf16 (16B) each
#pragma unroll
    for (int t = 0; t < N_ET; ++t) {
#pragma unroll
        for (int r = 0; r < 8; ++r)
            st[r + lhalf * 8][o0 + lmod] = (bf16_t)c[t][r];
        __syncthreads();
        *(uint4*)(T + ((size_t)t * N + m0 + srow) * DIM + schk * 8) =
            *(const uint4*)&st[srow][schk * 8];
        __syncthreads();
    }
}

// ------------------------------------------- a[dst] += T[etype, src]  (bf16 gather)
__global__ __launch_bounds__(256)
void edge_scatter_kernel(const int* __restrict__ src, const int* __restrict__ dst,
                         const int* __restrict__ ety, const bf16_t* __restrict__ T,
                         float* __restrict__ a, int E, int N) {
    long gid = (long)blockIdx.x * blockDim.x + threadIdx.x;   // E * 16 threads
    long e = gid >> 4;
    if (e >= E) return;
    int c8 = (int)(gid & 15) * 8;
    int s = src[e], d = dst[e], t = ety[e];
    v8bf v = *(const v8bf*)(T + ((size_t)t * N + (size_t)s) * DIM + c8);
    float* ap = a + (size_t)d * DIM + c8;
#pragma unroll
    for (int j = 0; j < 8; ++j)
        unsafeAtomicAdd(ap + j, (float)v[j]);   // native global_atomic_add_f32
}

// ------------- fused GRU: gi = a@Wih^T+bih, gh = h@Whh^T+bhh, gates, h update
__global__ __launch_bounds__(256)
void gru_fused_kernel(float* __restrict__ h, const float* __restrict__ a,
                      const bf16_t* __restrict__ Wih, const bf16_t* __restrict__ Whh,
                      const float* __restrict__ bih, const float* __restrict__ bhh,
                      int N) {
    __shared__ float gi[16][GDIM3];
    __shared__ float gh[16][GDIM3];

    const int m0    = blockIdx.x * 16;
    const int wave  = threadIdx.x >> 5;
    const int lane  = threadIdx.x & 31;
    const int lmod  = lane & 15;
    const int lhalf = lane >> 4;

    const float* __restrict__ arow = a + (size_t)(m0 + lmod) * DIM;
    const float* __restrict__ hrow = h + (size_t)(m0 + lmod) * DIM;

    // wave w owns output-col tiles {w, w+8, w+16} of the 24 tiles (384 cols)
    v8f ci[3], ch[3];
#pragma unroll
    for (int q = 0; q < 3; ++q) {
        int n  = (wave + 8 * q) * 16 + lmod;
        float b1 = bih[n], b2 = bhh[n];
#pragma unroll
        for (int r = 0; r < 8; ++r) { ci[q][r] = b1; ch[q][r] = b2; }
    }

#pragma unroll
    for (int kk = 0; kk < DIM; kk += 32) {
        v16bf aa = load_a_bf16(arow, kk + lhalf * 16);
        v16bf ah = load_a_bf16(hrow, kk + lhalf * 16);
        int kb2 = kk + lhalf * 16;
#pragma unroll
        for (int q = 0; q < 3; ++q) {
            int n = (wave + 8 * q) * 16 + lmod;
            v16bf bi = *(const v16bf*)(Wih + (size_t)n * DIM + kb2);
            v16bf bh = *(const v16bf*)(Whh + (size_t)n * DIM + kb2);
            ci[q] = __builtin_amdgcn_wmma_f32_16x16x32_bf16(false, aa, false, bi,
                                                            (short)0, ci[q], false, false);
            ch[q] = __builtin_amdgcn_wmma_f32_16x16x32_bf16(false, ah, false, bh,
                                                            (short)0, ch[q], false, false);
        }
    }

#pragma unroll
    for (int q = 0; q < 3; ++q) {
        int n = (wave + 8 * q) * 16 + lmod;
#pragma unroll
        for (int r = 0; r < 8; ++r) {
            int ml = r + lhalf * 8;
            gi[ml][n] = ci[q][r];
            gh[ml][n] = ch[q][r];
        }
    }
    __syncthreads();

    // 16 nodes x 128 dims = 2048 values; 256 threads x 8
#pragma unroll
    for (int it = 0; it < 8; ++it) {
        int idx = threadIdx.x + it * 256;
        int nl  = idx >> 7;
        int dc  = idx & 127;
        float ir = gi[nl][dc], iz = gi[nl][DIM + dc], in_ = gi[nl][2 * DIM + dc];
        float hr = gh[nl][dc], hz = gh[nl][DIM + dc], hn  = gh[nl][2 * DIM + dc];
        float rg = 1.0f / (1.0f + __expf(-(ir + hr)));
        float zg = 1.0f / (1.0f + __expf(-(iz + hz)));
        float ng = tanhf(in_ + rg * hn);
        size_t hi = (size_t)(m0 + nl) * DIM + dc;
        float hold = h[hi];
        h[hi] = (1.0f - zg) * ng + zg * hold;   // block owns rows m0..m0+15 exclusively
    }
}

// -------------------------------------------------- graph segment-sum + classifier
__global__ void graph_reduce_kernel(const float* __restrict__ h,
                                    const int* __restrict__ gids,
                                    float* __restrict__ gsum, int N) {
    long i = (long)blockIdx.x * blockDim.x + threadIdx.x;
    if (i >= (long)N * DIM) return;
    int n  = (int)(i >> 7);
    int dc = (int)(i & 127);
    unsafeAtomicAdd(&gsum[(size_t)gids[n] * DIM + dc], h[i]);
}

__global__ void logits_kernel(const float* __restrict__ gsum,
                              const float* __restrict__ cw,
                              const float* __restrict__ cb,
                              float* __restrict__ out, int B) {
    int g = threadIdx.x;
    if (g < B) {
        float s = 0.0f;
#pragma unroll 8
        for (int d = 0; d < DIM; ++d) s += gsum[(size_t)g * DIM + d] * cw[d];
        s += cb[0];
        out[g] = 1.0f / (1.0f + __expf(-s));
    }
}

// ------------------------------------------------------------------------- launch
extern "C" void kernel_launch(void* const* d_in, const int* in_sizes, int n_in,
                              void* d_out, int out_size, void* d_ws, size_t ws_size,
                              hipStream_t stream) {
    const float* features = (const float*)d_in[0];
    const int*   src      = (const int*)d_in[1];
    const int*   dst      = (const int*)d_in[2];
    const int*   ety      = (const int*)d_in[3];
    const int*   gids     = (const int*)d_in[4];
    const float* W_et     = (const float*)d_in[5];
    const float* b_et     = (const float*)d_in[6];
    const float* W_ih     = (const float*)d_in[7];
    const float* W_hh     = (const float*)d_in[8];
    const float* b_ih     = (const float*)d_in[9];
    const float* b_hh     = (const float*)d_in[10];
    const float* cls_w    = (const float*)d_in[11];
    const float* cls_b    = (const float*)d_in[12];

    const int N = in_sizes[0] / DIM;   // 50000 (multiple of 16)
    const int E = in_sizes[1];         // 500000
    const int B = out_size;            // 64 graphs

    // workspace layout (bf16 T first; ~103 MB total)
    bf16_t* T    = (bf16_t*)d_ws;                      // [4, N, 128] bf16
    float*  h    = (float*)(T + (size_t)N_ET * N * DIM); // [N, 128]  f32
    float*  a    = h + (size_t)N * DIM;                // [N, 128]    f32
    float*  gsum = a + (size_t)N * DIM;                // [B, 128]    f32
    bf16_t* dWe  = (bf16_t*)(gsum + (size_t)B * DIM);  // [4,128,128] bf16
    bf16_t* dWih = dWe + N_ET * DIM * DIM;             // [384,128]   bf16
    bf16_t* dWhh = dWih + GDIM3 * DIM;                 // [384,128]   bf16

    hipMemcpyAsync(h, features, (size_t)N * DIM * sizeof(float),
                   hipMemcpyDeviceToDevice, stream);
    cvt_weights_kernel<<<(N_ET * DIM * DIM + 255) / 256, 256, 0, stream>>>(
        W_et, W_ih, W_hh, dWe, dWih, dWhh);

    const int mtiles = N / 16;
    const long ndval = (long)N * DIM;

    for (int step = 0; step < 8; ++step) {
        etype_gemm_kernel<<<mtiles, 256, 0, stream>>>(h, dWe, b_et, T, N);
        zero_f32_kernel<<<(int)((ndval + 255) / 256), 256, 0, stream>>>(a, ndval);
        edge_scatter_kernel<<<(int)(((long)E * 16 + 255) / 256), 256, 0, stream>>>(
            src, dst, ety, T, a, E, N);
        gru_fused_kernel<<<mtiles, 256, 0, stream>>>(h, a, dWih, dWhh, b_ih, b_hh, N);
    }

    zero_f32_kernel<<<(int)(((long)B * DIM + 255) / 256), 256, 0, stream>>>(
        gsum, (long)B * DIM);
    graph_reduce_kernel<<<(int)((ndval + 255) / 256), 256, 0, stream>>>(h, gids, gsum, N);
    logits_kernel<<<1, B > 0 ? B : 1, 0, stream>>>(gsum, cls_w, cls_b, (float*)d_out, B);
}
